// GCNTeacher_23957327577190
// MI455X (gfx1250) — matrix operand order repaired and checked by reading, not compiled, not run on hardware
//
#include <hip/hip_runtime.h>
#include <cstdint>

typedef __attribute__((ext_vector_type(16))) __bf16 v16bf;
typedef __attribute__((ext_vector_type(2)))  __bf16 v2bf;
typedef __attribute__((ext_vector_type(8)))  float  v8f;

#define EPSV 1e-5f

// Native f32 -> bf16 (RNE). Prefer the packed hardware convert when declared.
#if defined(__has_builtin) && __has_builtin(__builtin_amdgcn_cvt_pk_bf16_f32)
static __device__ __forceinline__ v2bf f2bf2(float lo, float hi) {
  return __builtin_amdgcn_cvt_pk_bf16_f32(lo, hi);
}
#else
static __device__ __forceinline__ v2bf f2bf2(float lo, float hi) {
  v2bf r;
  r.x = (__bf16)lo;   // fptrunc, RNE; gfx1250 lowers to hw convert
  r.y = (__bf16)hi;
  return r;
}
#endif

static __device__ __forceinline__ __bf16 f2bf(float f) { return (__bf16)f; }

// ---------------- degree / norm ----------------
__global__ void k_deg(const int* __restrict__ src, const int* __restrict__ dst,
                      float* __restrict__ odeg, float* __restrict__ ideg, int E) {
  int e = blockIdx.x * blockDim.x + threadIdx.x;
  if (e < E) {
    atomicAdd(&odeg[src[e]], 1.0f);
    atomicAdd(&ideg[dst[e]], 1.0f);
  }
}

__global__ void k_norm(const float* __restrict__ odeg, const float* __restrict__ ideg,
                       float* __restrict__ onorm, float* __restrict__ inorm, int n) {
  int i = blockIdx.x * blockDim.x + threadIdx.x;
  if (i < n) {
    onorm[i] = rsqrtf(fmaxf(odeg[i], 1.0f));
    inorm[i] = rsqrtf(fmaxf(ideg[i], 1.0f));
  }
}

// ------------- pack W[K][ncols] (f32 row-major) into WMMA B-fragment order (bf16) -------------
// packed idx = ((tc*nkc + kc)*32 + lane)*16 + e  ;  value = W[kc*32 + (e<8 ? (lane>>4)*8 + e : 16 + (lane>>4)*8 + (e-8))][tc*16 + (lane&15)]
__global__ void k_pack_w(const float* __restrict__ W, __bf16* __restrict__ Wp,
                         int K, int ncols, int NT) {
  int nkc = K / 32;
  int total = NT * nkc * 512;
  int idx = blockIdx.x * blockDim.x + threadIdx.x;
  if (idx >= total) return;
  int e  = idx & 15;
  int l  = (idx >> 4) & 31;
  int kc = (idx >> 9) % nkc;
  int tc = (idx >> 9) / nkc;
  int k  = kc * 32 + ((e < 8) ? ((l >> 4) * 8 + e) : (16 + (l >> 4) * 8 + (e - 8)));
  int c  = tc * 16 + (l & 15);
  float v = (c < ncols) ? W[(size_t)k * ncols + c] : 0.0f;
  Wp[idx] = f2bf(v);
}

// ------------- WMMA GEMM (bf16 inputs, f32 accumulate), fused src-norm scale -------------
// One wave computes one 16x16 output tile with K=KDIM via KDIM/32 chained WMMAs.
// block = (32, NT): threadIdx.y selects the output column tile.
template <int KDIM>
__global__ void k_gemm_scale(const float* __restrict__ Hin, const __bf16* __restrict__ Wp,
                             const float* __restrict__ onorm, float* __restrict__ Hout,
                             int nrows, int ldout) {
  constexpr int NKC = KDIM / 32;
  const int lane = threadIdx.x;           // 0..31
  const int tc   = threadIdx.y;           // output column tile
  const int m    = lane & 15;
  const int half = lane >> 4;
  const int trow = blockIdx.x;
  int row = trow * 16 + m;
  if (row > nrows - 1) row = nrows - 1;   // clamp; N%16==0 in practice
  const float* base = Hin + (size_t)row * KDIM;

  v8f acc = {};
#pragma unroll
  for (int kc = 0; kc < NKC; ++kc) {
    const float* ar = base + kc * 32;
    // A-fragment layout (16-bit A, 16x32): elems 0..7 -> K = half*8 + e ; elems 8..15 -> K = 16 + half*8 + (e-8)
    float4 q0 = *reinterpret_cast<const float4*>(ar + half * 8);
    float4 q1 = *reinterpret_cast<const float4*>(ar + half * 8 + 4);
    float4 q2 = *reinterpret_cast<const float4*>(ar + 16 + half * 8);
    float4 q3 = *reinterpret_cast<const float4*>(ar + 20 + half * 8);
    v16bf a;
    v2bf p;
    p = f2bf2(q0.x, q0.y); a[0]  = p.x; a[1]  = p.y;
    p = f2bf2(q0.z, q0.w); a[2]  = p.x; a[3]  = p.y;
    p = f2bf2(q1.x, q1.y); a[4]  = p.x; a[5]  = p.y;
    p = f2bf2(q1.z, q1.w); a[6]  = p.x; a[7]  = p.y;
    p = f2bf2(q2.x, q2.y); a[8]  = p.x; a[9]  = p.y;
    p = f2bf2(q2.z, q2.w); a[10] = p.x; a[11] = p.y;
    p = f2bf2(q3.x, q3.y); a[12] = p.x; a[13] = p.y;
    p = f2bf2(q3.z, q3.w); a[14] = p.x; a[15] = p.y;
    // B-fragment: pre-packed contiguous 32 bytes per lane
    v16bf b = *reinterpret_cast<const v16bf*>(Wp + (((size_t)tc * NKC + kc) * 32 + lane) * 16);
    acc = __builtin_amdgcn_wmma_f32_16x16x32_bf16(false, a, false, b, (short)0, acc, false, false);
  }

  const int n = tc * 16 + m;
#pragma unroll
  for (int r = 0; r < 8; ++r) {
    int rr = trow * 16 + half * 8 + r;    // C/D layout: VGPR r -> M = half*8 + r
    if (rr < nrows) Hout[(size_t)rr * ldout + n] = acc[r] * onorm[rr];
  }
}

// ------------- edge scatter-add (agg fits in 192MB L2 -> L2-resident float atomics) -------------
__global__ void k_scatter(const float* __restrict__ Hsrc, const int* __restrict__ src,
                          const int* __restrict__ dst, float* __restrict__ Agg,
                          int E, int F, int chunks) {
  long long tid = (long long)blockIdx.x * blockDim.x + threadIdx.x;
  long long total = (long long)E * chunks;
  if (tid >= total) return;
  int e = (int)(tid / chunks);
  int c = (int)(tid % chunks) * 4;
  int s = src[e], d = dst[e];
  float4 v = *reinterpret_cast<const float4*>(Hsrc + (size_t)s * F + c);
  float* p = Agg + (size_t)d * F + c;
  atomicAdd(p + 0, v.x);
  atomicAdd(p + 1, v.y);
  atomicAdd(p + 2, v.z);
  atomicAdd(p + 3, v.w);
}

// ------------- dst-norm scale + bias (in place) -------------
__global__ void k_postagg(float* __restrict__ H, const float* __restrict__ inorm,
                          const float* __restrict__ bias, int nrows, int F) {
  long long idx = (long long)blockIdx.x * blockDim.x + threadIdx.x;
  if (idx >= (long long)nrows * F) return;
  int r = (int)(idx / F);
  int c = (int)(idx % F);
  H[idx] = H[idx] * inorm[r] + bias[c];
}

// ------------- per-column mean / mean-square accumulation (F == 128, blockDim == 256) -------------
__global__ void k_colstats(const float* __restrict__ H, int nrows, int F,
                           float* __restrict__ colsum, float* __restrict__ colsq) {
  int c   = threadIdx.x % F;          // F == 128
  int sub = threadIdx.x / F;          // 0..1
  int rpb = blockDim.x / F;           // 2
  float s = 0.0f, q = 0.0f;
  for (int r = blockIdx.x * rpb + sub; r < nrows; r += gridDim.x * rpb) {
    float v = H[(size_t)r * F + c];
    s += v; q += v * v;
  }
  __shared__ float ss[256], qq[256];
  ss[threadIdx.x] = s; qq[threadIdx.x] = q;
  __syncthreads();
  if (sub == 0) {
    s += ss[threadIdx.x + F];
    q += qq[threadIdx.x + F];
    atomicAdd(&colsum[c], s);
    atomicAdd(&colsq[c], q);
  }
}

__global__ void k_bn_relu(float* __restrict__ H, const float* __restrict__ colsum,
                          const float* __restrict__ colsq, const float* __restrict__ gamma,
                          const float* __restrict__ beta, int nrows, int F) {
  long long idx = (long long)blockIdx.x * blockDim.x + threadIdx.x;
  if (idx >= (long long)nrows * F) return;
  int c = (int)(idx % F);
  float invn = 1.0f / (float)nrows;
  float mu = colsum[c] * invn;
  float var = colsq[c] * invn - mu * mu;
  float v = (H[idx] - mu) * rsqrtf(var + EPSV) * gamma[c] + beta[c];
  H[idx] = v > 0.0f ? v : 0.0f;
}

// ------------- final layer: dst-norm + bias, 48-wide padded agg -> 40-wide output -------------
__global__ void k_postagg_out(const float* __restrict__ Agg48, const float* __restrict__ inorm,
                              const float* __restrict__ b2, float* __restrict__ out, int nrows) {
  long long idx = (long long)blockIdx.x * blockDim.x + threadIdx.x;
  if (idx >= (long long)nrows * 40) return;
  int r = (int)(idx / 40);
  int c = (int)(idx % 40);
  out[idx] = Agg48[(size_t)r * 48 + c] * inorm[r] + b2[c];
}

extern "C" void kernel_launch(void* const* d_in, const int* in_sizes, int n_in,
                              void* d_out, int out_size, void* d_ws, size_t ws_size,
                              hipStream_t stream) {
  const float* feat = (const float*)d_in[0];
  const int*   src  = (const int*)d_in[1];
  const int*   dst  = (const int*)d_in[2];
  const float* W0   = (const float*)d_in[3];
  const float* b0   = (const float*)d_in[4];
  const float* g0   = (const float*)d_in[5];
  const float* be0  = (const float*)d_in[6];
  const float* W1   = (const float*)d_in[7];
  const float* b1   = (const float*)d_in[8];
  const float* g1   = (const float*)d_in[9];
  const float* be1  = (const float*)d_in[10];
  const float* W2   = (const float*)d_in[11];
  const float* b2   = (const float*)d_in[12];

  const int Nn = in_sizes[0] / 128;   // 100000
  const int Ee = in_sizes[1];         // 1600000

  // workspace carve-out (256B aligned)
  char* base = (char*)d_ws;
  size_t off = 0;
  auto carve = [&](size_t bytes) -> char* {
    char* p = base + off;
    off = (off + bytes + 255) & ~(size_t)255;
    return p;
  };
  const size_t big = (size_t)Nn * 128 * sizeof(float);
  float*  bufA   = (float*)carve(big);
  float*  bufB   = (float*)carve(big);
  float*  odeg   = (float*)carve((size_t)Nn * 4);
  float*  ideg   = (float*)carve((size_t)Nn * 4);
  float*  onorm  = (float*)carve((size_t)Nn * 4);
  float*  inorm  = (float*)carve((size_t)Nn * 4);
  __bf16* Wp0    = (__bf16*)carve(128 * 128 * 2);
  __bf16* Wp1    = (__bf16*)carve(128 * 128 * 2);
  __bf16* Wp2    = (__bf16*)carve(128 * 48 * 2);
  float*  colsum = (float*)carve(128 * 4);
  float*  colsq  = (float*)carve(128 * 4);
  if (off > ws_size) return;  // workspace too small; bail safely

  const int TB = 256;
  const int rowTiles = (Nn + 15) / 16;

  // 1) degrees + norms
  hipMemsetAsync(odeg, 0, (size_t)Nn * 4, stream);
  hipMemsetAsync(ideg, 0, (size_t)Nn * 4, stream);
  k_deg<<<(Ee + TB - 1) / TB, TB, 0, stream>>>(src, dst, odeg, ideg, Ee);
  k_norm<<<(Nn + TB - 1) / TB, TB, 0, stream>>>(odeg, ideg, onorm, inorm, Nn);

  // 2) pack weights into WMMA B-fragment order
  k_pack_w<<<(8 * 4 * 512 + TB - 1) / TB, TB, 0, stream>>>(W0, Wp0, 128, 128, 8);
  k_pack_w<<<(8 * 4 * 512 + TB - 1) / TB, TB, 0, stream>>>(W1, Wp1, 128, 128, 8);
  k_pack_w<<<(3 * 4 * 512 + TB - 1) / TB, TB, 0, stream>>>(W2, Wp2, 128, 40, 3);

  long long tot128 = (long long)Nn * 128;
  long long sc128  = (long long)Ee * 32;
  long long sc48   = (long long)Ee * 12;

  // ---- layer 0 ----
  k_gemm_scale<128><<<dim3(rowTiles), dim3(32, 8), 0, stream>>>(feat, Wp0, onorm, bufA, Nn, 128);
  hipMemsetAsync(bufB, 0, big, stream);
  k_scatter<<<(int)((sc128 + TB - 1) / TB), TB, 0, stream>>>(bufA, src, dst, bufB, Ee, 128, 32);
  k_postagg<<<(int)((tot128 + TB - 1) / TB), TB, 0, stream>>>(bufB, inorm, b0, Nn, 128);
  hipMemsetAsync(colsum, 0, 128 * 4, stream);
  hipMemsetAsync(colsq, 0, 128 * 4, stream);
  k_colstats<<<256, 256, 0, stream>>>(bufB, Nn, 128, colsum, colsq);
  k_bn_relu<<<(int)((tot128 + TB - 1) / TB), TB, 0, stream>>>(bufB, colsum, colsq, g0, be0, Nn, 128);

  // ---- layer 1 ----
  k_gemm_scale<128><<<dim3(rowTiles), dim3(32, 8), 0, stream>>>(bufB, Wp1, onorm, bufA, Nn, 128);
  hipMemsetAsync(bufB, 0, big, stream);
  k_scatter<<<(int)((sc128 + TB - 1) / TB), TB, 0, stream>>>(bufA, src, dst, bufB, Ee, 128, 32);
  k_postagg<<<(int)((tot128 + TB - 1) / TB), TB, 0, stream>>>(bufB, inorm, b1, Nn, 128);
  hipMemsetAsync(colsum, 0, 128 * 4, stream);
  hipMemsetAsync(colsq, 0, 128 * 4, stream);
  k_colstats<<<256, 256, 0, stream>>>(bufB, Nn, 128, colsum, colsq);
  k_bn_relu<<<(int)((tot128 + TB - 1) / TB), TB, 0, stream>>>(bufB, colsum, colsq, g1, be1, Nn, 128);

  // ---- layer 2 (NCLS=40 padded to 48 = 3 column tiles; pad cols are exactly zero) ----
  k_gemm_scale<128><<<dim3(rowTiles), dim3(32, 3), 0, stream>>>(bufB, Wp2, onorm, bufA, Nn, 48);
  hipMemsetAsync(bufB, 0, (size_t)Nn * 48 * sizeof(float), stream);
  k_scatter<<<(int)((sc48 + TB - 1) / TB), TB, 0, stream>>>(bufA, src, dst, bufB, Ee, 48, 12);
  long long totOut = (long long)Nn * 40;
  k_postagg_out<<<(int)((totOut + TB - 1) / TB), TB, 0, stream>>>(bufB, inorm, b2, (float*)d_out, Nn);
}
